// LSTMWithHiddenAttention_69191923139129
// MI455X (gfx1250) — compile-verified, matrix operand order checked by reading, not run on hardware
//
#include <hip/hip_runtime.h>

typedef unsigned short u16;
typedef __attribute__((ext_vector_type(16))) __bf16 v16bf;
typedef __attribute__((ext_vector_type(8)))  float  v8f;

#define B_   16
#define S_   256
#define I_   128
#define HS_  256
#define R_   8
#define DA_  64
#define NG_  1024            // 4*HS
#define KHH_ 2048            // R*HS

// float -> bf16 round-to-nearest-even
__device__ __forceinline__ u16 f2bf(float f) {
  unsigned int x = __float_as_uint(f);
  unsigned int r = x + 0x7FFFu + ((x >> 16) & 1u);
  return (u16)(r >> 16);
}

// Load one wave32 bf16 A/B fragment (16x32, row-major, row fixed per lane).
// elements [0..7]  = K kk+hi*8    ... contiguous -> one b128
// elements [8..15] = K kk+16+hi*8 ... contiguous -> one b128
__device__ __forceinline__ v16bf load_frag(const u16* rowbase, int kk, int hi) {
  union { uint4 q; __bf16 b[8]; } lo, hw;
  lo.q = *(const uint4*)(rowbase + kk + hi * 8);
  hw.q = *(const uint4*)(rowbase + kk + 16 + hi * 8);
  v16bf f;
#pragma unroll
  for (int e = 0; e < 8; ++e) { f[e] = lo.b[e]; f[8 + e] = hw.b[e]; }
  return f;
}

// ---------------- one-time prep: bf16 casts + weight transposes + zero c ----------------
__global__ void prep_kernel(const float* __restrict__ x, const float* __restrict__ wih,
                            const float* __restrict__ whh, const float* __restrict__ W1,
                            float* __restrict__ c, u16* __restrict__ xbf,
                            u16* __restrict__ wih_t, u16* __restrict__ whh_t,
                            u16* __restrict__ w1_t) {
  int idx0 = blockIdx.x * blockDim.x + threadIdx.x;
  int stride = gridDim.x * blockDim.x;
  for (int i = idx0; i < B_ * HS_; i += stride) c[i] = 0.0f;
  for (int i = idx0; i < B_ * S_ * I_; i += stride) xbf[i] = f2bf(x[i]);
  for (int i = idx0; i < NG_ * I_; i += stride) {        // wih_t[n][k] = wih[k][n]
    int n = i / I_, k = i % I_;
    wih_t[i] = f2bf(wih[k * NG_ + n]);
  }
  for (int i = idx0; i < NG_ * KHH_; i += stride) {      // whh_t[n][k] = whh[k][n]
    int n = i / KHH_, k = i % KHH_;
    whh_t[i] = f2bf(whh[k * NG_ + n]);
  }
  for (int i = idx0; i < DA_ * HS_; i += stride) {       // w1_t[n][k] = W1[k][n]
    int n = i / HS_, k = i % HS_;
    w1_t[i] = f2bf(W1[k * DA_ + n]);
  }
}

// ---------------- per-step attention: softmax over score rows < t, M = attn^T Hbuf ------
__global__ void att_kernel(const float* __restrict__ scoreBuf,
                           const float* __restrict__ Hbuf,   // = d_out hidden_seq region
                           u16* __restrict__ hBar, int t) {
  int b = blockIdx.x;
  int tid = threadIdx.x;
  if (t == 0) {                    // reference attends a single zero row -> hBar = 0
#pragma unroll
    for (int r = 0; r < R_; ++r) hBar[b * KHH_ + r * HS_ + tid] = 0;
    return;
  }
  __shared__ float attnW[S_ * R_];
  __shared__ float invS[R_];
  int wave = tid >> 5, lane = tid & 31;
  {                                // wave w owns head r = w (8 waves == R heads)
    int r = wave;
    const float* sc = scoreBuf + (size_t)b * S_ * R_ + r;
    float m = -1e30f;
    for (int s = lane; s < t; s += 32) m = fmaxf(m, sc[s * R_]);
#pragma unroll
    for (int off = 16; off; off >>= 1) m = fmaxf(m, __shfl_xor(m, off, 32));
    float sum = 0.0f;
    for (int s = lane; s < t; s += 32) {
      float e = __expf(sc[s * R_] - m);
      attnW[s * R_ + r] = e;
      sum += e;
    }
#pragma unroll
    for (int off = 16; off; off >>= 1) sum += __shfl_xor(sum, off, 32);
    if (lane == 0) invS[r] = 1.0f / sum;
  }
  __syncthreads();
  int j = tid;                      // one hidden column per thread
  float acc[R_];
#pragma unroll
  for (int r = 0; r < R_; ++r) acc[r] = 0.0f;
  const float* Hb = Hbuf + (size_t)b * S_ * HS_ + j;
  for (int s = 0; s < t; ++s) {
    float hv = Hb[(size_t)s * HS_];
    const float* w = &attnW[s * R_];
#pragma unroll
    for (int r = 0; r < R_; ++r) acc[r] += w[r] * hv;
  }
#pragma unroll
  for (int r = 0; r < R_; ++r)
    hBar[b * KHH_ + r * HS_ + j] = f2bf(acc[r] * invS[r]);
}

// ---------------- per-step gates GEMM: [16 x 2176] @ [2176 x 1024] via bf16 WMMA --------
__global__ void gates_kernel(const u16* __restrict__ xbf, const u16* __restrict__ hBar,
                             const u16* __restrict__ wih_t, const u16* __restrict__ whh_t,
                             const float* __restrict__ bias, float* __restrict__ gates,
                             int t) {
  __shared__ float red[4][32][8];
  int ntile = blockIdx.x;                     // 64 tiles of 16 gate columns
  // wave index is wave-uniform: force scalar so loop control is SALU, not exec-mask
  int wave = __builtin_amdgcn_readfirstlane(threadIdx.x >> 5);
  int lane = threadIdx.x & 31;
  int row = lane & 15, hi = lane >> 4;
  int col = ntile * 16 + row;
  const u16* arow_x = xbf + ((size_t)row * S_ + t) * I_;   // A row = batch
  const u16* arow_h = hBar + (size_t)row * KHH_;
  const u16* brow_x = wih_t + (size_t)col * I_;            // B via transposed weights
  const u16* brow_h = whh_t + (size_t)col * KHH_;
  v8f acc = {};
  // software pipeline: x-part chunk (chunk index == wave) is the prologue
  v16bf a  = load_frag(arow_x, wave * 32, hi);
  v16bf bf = load_frag(brow_x, wave * 32, hi);
  // hBar-part: exactly 16 K-chunks per wave (static trip count so unroll-2 can
  // alternate register sets and kill the rotation copies)
#pragma unroll 2
  for (int i = 0; i < 16; ++i) {
    int kk = wave * 32 + i * 128;
    v16bf an = load_frag(arow_h, kk, hi);            // next fragments in flight ...
    v16bf bn = load_frag(brow_h, kk, hi);
    acc = __builtin_amdgcn_wmma_f32_16x16x32_bf16(false, a, false, bf,
                                                  (short)0, acc, false, false);
    a = an; bf = bn;                                 // ... while current WMMA issues
  }
  acc = __builtin_amdgcn_wmma_f32_16x16x32_bf16(false, a, false, bf,
                                                (short)0, acc, false, false);
#pragma unroll
  for (int v = 0; v < 8; ++v) red[wave][lane][v] = acc[v];
  __syncthreads();
  if (wave == 0) {
    float bv = bias[col];
#pragma unroll
    for (int v = 0; v < 8; ++v) {             // C layout: b = v + 8*hi, n = col
      float s = red[0][lane][v] + red[1][lane][v] + red[2][lane][v] + red[3][lane][v];
      gates[(v + 8 * hi) * NG_ + col] = s + bv;
    }
  }
}

// ---------------- per-step cell update + score row t (h@W1 via WMMA, then @W2) ----------
__global__ void cell_kernel(const float* __restrict__ gates, float* __restrict__ c,
                            float* __restrict__ out, const u16* __restrict__ w1_t,
                            const float* __restrict__ b1, const float* __restrict__ W2,
                            const float* __restrict__ b2, float* __restrict__ scoreBuf,
                            int t) {
  __shared__ __align__(16) u16 hbf[B_ * HS_];
  __shared__ float uLds[B_ * DA_];
  int tid = threadIdx.x;
  float* hseq = out;                          // [B][S][HS]
  float* hT = out + (size_t)B_ * S_ * HS_;
  float* cT = hT + B_ * HS_;
  for (int u = tid; u < B_ * HS_; u += blockDim.x) {
    int b = u >> 8, j = u & 255;
    float gi = gates[b * NG_ + j];
    float gf = gates[b * NG_ + 256 + j];
    float gg = gates[b * NG_ + 512 + j];
    float go = gates[b * NG_ + 768 + j];
    float it = 1.0f / (1.0f + __expf(-gi));
    float ft = 1.0f / (1.0f + __expf(-gf));
    float gt = tanhf(gg);
    float ot = 1.0f / (1.0f + __expf(-go));
    float cn = ft * c[u] + it * gt;
    float h = ot * tanhf(cn);
    c[u] = cn;
    hseq[((size_t)b * S_ + t) * HS_ + j] = h;
    hbf[u] = f2bf(h);
    if (t == S_ - 1) { hT[u] = h; cT[u] = cn; }
  }
  __syncthreads();
  int wave = __builtin_amdgcn_readfirstlane(threadIdx.x >> 5);
  int lane = tid & 31;
  if (wave < 4) {                             // [16x256] @ [256x64]: 4 N-tiles, K=256
    int row = lane & 15, hi = lane >> 4;
    int col = wave * 16 + row;
    const u16* arow = &hbf[row * HS_];
    const u16* brow = w1_t + (size_t)col * HS_;
    v8f acc = {};
#pragma unroll
    for (int ck = 0; ck < 8; ++ck) {
      v16bf a  = load_frag(arow, ck * 32, hi);
      v16bf bb = load_frag(brow, ck * 32, hi);
      acc = __builtin_amdgcn_wmma_f32_16x16x32_bf16(false, a, false, bb,
                                                    (short)0, acc, false, false);
    }
#pragma unroll
    for (int v = 0; v < 8; ++v) {
      int b = v + 8 * hi;
      int n = wave * 16 + (lane & 15);
      uLds[b * DA_ + n] = tanhf(acc[v] + b1[n]);
    }
  }
  __syncthreads();
  if (tid < B_ * R_) {                        // tiny [16x64]@[64x8]
    int b = tid >> 3, r = tid & 7;
    float s = b2[r];
#pragma unroll
    for (int d = 0; d < DA_; ++d) s += uLds[b * DA_ + d] * W2[d * R_ + r];
    scoreBuf[((size_t)b * S_ + t) * R_ + r] = s;
  }
}

extern "C" void kernel_launch(void* const* d_in, const int* in_sizes, int n_in,
                              void* d_out, int out_size, void* d_ws, size_t ws_size,
                              hipStream_t stream) {
  (void)in_sizes; (void)n_in; (void)out_size; (void)ws_size;
  const float* x    = (const float*)d_in[0];
  const float* wih  = (const float*)d_in[1];
  const float* whh  = (const float*)d_in[2];
  const float* bias = (const float*)d_in[3];
  const float* W1   = (const float*)d_in[4];
  const float* b1   = (const float*)d_in[5];
  const float* W2   = (const float*)d_in[6];
  const float* b2   = (const float*)d_in[7];
  float* out = (float*)d_out;

  char* ws = (char*)d_ws;
  float* c_ws     = (float*)ws; ws += (size_t)B_ * HS_ * 4;
  float* gates    = (float*)ws; ws += (size_t)B_ * NG_ * 4;
  float* scoreBuf = (float*)ws; ws += (size_t)B_ * S_ * R_ * 4;
  u16*   hBar     = (u16*)ws;   ws += (size_t)B_ * KHH_ * 2;
  u16*   xbf      = (u16*)ws;   ws += (size_t)B_ * S_ * I_ * 2;
  u16*   wih_t    = (u16*)ws;   ws += (size_t)NG_ * I_ * 2;
  u16*   whh_t    = (u16*)ws;   ws += (size_t)NG_ * KHH_ * 2;
  u16*   w1_t     = (u16*)ws;   // DA_*HS_*2 bytes

  prep_kernel<<<2048, 256, 0, stream>>>(x, wih, whh, W1, c_ws, xbf, wih_t, whh_t, w1_t);

  for (int t = 0; t < S_; ++t) {
    att_kernel<<<B_, 256, 0, stream>>>(scoreBuf, out, hBar, t);
    gates_kernel<<<NG_ / 16, 128, 0, stream>>>(xbf, hBar, wih_t, whh_t, bias, gates, t);
    cell_kernel<<<1, 512, 0, stream>>>(gates, c_ws, out, w1_t, b1, W2, b2, scoreBuf, t);
  }
}
